// RetrievalDatabase_57028575756700
// MI455X (gfx1250) — matrix operand order repaired and checked by reading, not compiled, not run on hardware
//
#include <hip/hip_runtime.h>
#include <math.h>

// ---------------------------------------------------------------------------
// Types for CDNA5 WMMA (wave32, v_wmma_f32_16x16x32_bf16)
// ---------------------------------------------------------------------------
typedef __bf16 bf16;
typedef bf16  bf16x8 __attribute__((ext_vector_type(8)));
typedef bf16  v16bf  __attribute__((ext_vector_type(16)));
typedef float v8f    __attribute__((ext_vector_type(8)));

// Problem constants (from the reference)
#define BQ     32
#define RK     4
#define NDB    4096
#define TMAXC  196
#define DMC    263
#define DMP    288      // DM padded to multiple of 32 for the K loop
#define DC     512
#define TCLIPC 77
#define HC     8
#define DHC    64
#define DFFC   1024
#define BRC    (BQ*RK)          // 128
#define M1C    (BRC*TMAXC)      // 25088 tokens (motion)
#define M2C    (BRC*TCLIPC)     // 9856 tokens (text)

__device__ __forceinline__ float gelu_exact(float x) {
    return 0.5f * x * (1.0f + erff(x * 0.70710678118654752f));
}

// ---------------------------------------------------------------------------
// Retrieval score: score[b,n] = cos(q_b, d_n) * exp(-0.1 * rel)
// ---------------------------------------------------------------------------
__global__ void score_kernel(const float* __restrict__ qt, const float* __restrict__ dbt,
                             const int* __restrict__ lengths, const int* __restrict__ mlen,
                             float* __restrict__ scores) {
    int b = blockIdx.x;
    int n = blockIdx.y * blockDim.x + threadIdx.x;   // grid.y*block = 4096
    const float* q = qt + (size_t)b * DC;
    const float* d = dbt + (size_t)n * DC;
    float dot = 0.f, qq = 0.f, dd = 0.f;
    for (int c = 0; c < DC; c += 4) {
        float4 qv = *(const float4*)(q + c);
        float4 dv = *(const float4*)(d + c);
        dot += qv.x*dv.x + qv.y*dv.y + qv.z*dv.z + qv.w*dv.w;
        qq  += qv.x*qv.x + qv.y*qv.y + qv.z*qv.z + qv.w*qv.w;
        dd  += dv.x*dv.x + dv.y*dv.y + dv.z*dv.z + dv.w*dv.w;
    }
    float sem = dot * rsqrtf(qq) * rsqrtf(dd);
    float len = (float)lengths[b];
    float a   = fabsf((float)mlen[n] - len);
    float rel = a / fmaxf(a, len);
    scores[(size_t)b * NDB + n] = sem * expf(-0.1f * rel);
}

// Sequential 4-way argmax per query (ties -> lowest index, matching lax.top_k)
__global__ void topk_kernel(float* __restrict__ scores, int* __restrict__ idx) {
    int b = blockIdx.x, tid = threadIdx.x;
    __shared__ float sv[256];
    __shared__ int   si[256];
    float* s = scores + (size_t)b * NDB;
    for (int r = 0; r < RK; ++r) {
        float bv = -INFINITY; int bi = 0x7fffffff;
        for (int n = tid; n < NDB; n += 256) {
            float v = s[n];
            if (v > bv || (v == bv && n < bi)) { bv = v; bi = n; }
        }
        sv[tid] = bv; si[tid] = bi; __syncthreads();
        for (int off = 128; off > 0; off >>= 1) {
            if (tid < off) {
                float ov = sv[tid+off]; int oi = si[tid+off];
                if (ov > sv[tid] || (ov == sv[tid] && oi < si[tid])) { sv[tid] = ov; si[tid] = oi; }
            }
            __syncthreads();
        }
        if (tid == 0) { idx[b*RK + r] = si[0]; s[si[0]] = -INFINITY; }
        __syncthreads();
    }
}

// ---------------------------------------------------------------------------
// Weight prep: W[K,N] f32 row-major -> Wt[N,Kp] bf16 (B^T, K zero-padded)
// ---------------------------------------------------------------------------
__global__ void convw_kernel(const float* __restrict__ W, bf16* __restrict__ Wt,
                             int K, int Nn, int Kp) {
    long total = (long)Nn * Kp;
    for (long e = blockIdx.x * (long)blockDim.x + threadIdx.x; e < total;
         e += (long)gridDim.x * blockDim.x) {
        int n = (int)(e / Kp), k = (int)(e % Kp);
        Wt[e] = (k < K) ? (bf16)W[(size_t)k * Nn + n] : (bf16)0.0f;
    }
}

// ---------------------------------------------------------------------------
// Gather kernels
// ---------------------------------------------------------------------------
__global__ void gather_motion_kernel(const float* __restrict__ dbm, const int* __restrict__ idx,
                                     bf16* __restrict__ Xm) {
    int row = blockIdx.x;                 // 0..M1-1
    int br = row / TMAXC, tt = row % TMAXC;
    int ii = idx[br];
    const float* src = dbm + ((size_t)ii * TMAXC + tt) * DMC;
    bf16* dst = Xm + (size_t)row * DMP;
    for (int c = threadIdx.x; c < DMP; c += blockDim.x)
        dst[c] = (c < DMC) ? (bf16)src[c] : (bf16)0.0f;
}

__global__ void gather_text_kernel(const float* __restrict__ dbc, const int* __restrict__ idx,
                                   float* __restrict__ t, bf16* __restrict__ tb) {
    int row = blockIdx.x;                 // 0..M2-1
    int br = row / TCLIPC, tt = row % TCLIPC;
    int ii = idx[br];
    const float* src = dbc + ((size_t)ii * TCLIPC + tt) * DC;
    for (int c = threadIdx.x; c < DC; c += blockDim.x) {
        float v = src[c];
        t [(size_t)row * DC + c] = v;
        tb[(size_t)row * DC + c] = (bf16)v;
    }
}

// ---------------------------------------------------------------------------
// Generic bf16 WMMA GEMM: C[M,N] = A[M,K] * Bt[N,K]^T (+bias +pos +residual, GELU)
//
// Workgroup tile 128x128 (8 waves), per-wave tile 32x64:
//   - 2 A fragments + 4 B fragments -> 8 WMMAs per K=32 step (1.5 b128 loads/WMMA)
//   - 8 independent accumulator chains keep the matrix pipe busy
//   - WGP$ dedupes A across the 2 N-waves and B across the 4 M-waves
//
// Fragment layouts per CDNA5 ISA 7.12.2:
//   A 16x32: lane L holds row (L&15); two contiguous 16B chunks at
//            k + 8*(L>>4) and +16 (row-major b128 loads)
//   B 32x16: lane L holds col (L&15); two contiguous 16B chunks at
//            k + 8*(L>>4) and +16 (same K order as A)
//   C/D:     reg r: m = r + 8*(L>>4), n = (L&15)
// ---------------------------------------------------------------------------
__device__ __forceinline__ v16bf ldfragA(const bf16* p) {
    bf16x8 lo = *(const bf16x8*)p;
    bf16x8 hi = *(const bf16x8*)(p + 16);
    v16bf r;
#pragma unroll
    for (int i = 0; i < 8; ++i) { r[i] = lo[i]; r[i + 8] = hi[i]; }
    return r;
}
__device__ __forceinline__ v16bf ldfragB(const bf16* p) {
    bf16x8 lo = *(const bf16x8*)p;
    bf16x8 hi = *(const bf16x8*)(p + 16);
    v16bf r;
#pragma unroll
    for (int i = 0; i < 8; ++i) { r[i] = lo[i]; r[i + 8] = hi[i]; }
    return r;
}

__global__ void __launch_bounds__(256, 1)
gemm_bf16_kernel(const bf16* __restrict__ A, int lda,
                 const bf16* __restrict__ Bt, int ldb,
                 const float* __restrict__ bias,
                 const float* __restrict__ resid,
                 const float* __restrict__ pose, int posT,
                 float* __restrict__ outF, bf16* __restrict__ outB,
                 int M, int Nn, int K, int do_gelu) {
    const int lane = threadIdx.x & 31;
    const int wave = threadIdx.x >> 5;        // 8 waves
    const int wm = wave & 3;                  // 4 M-waves * 32 rows
    const int wn = wave >> 2;                 // 2 N-waves * 64 cols
    const int mBase = blockIdx.x * 128 + wm * 32;
    const int nBase = blockIdx.y * 128 + wn * 64;
    if (mBase >= M || nBase >= Nn) return;

    const bf16* aRow0 = A  + (size_t)(mBase + (lane & 15)) * lda + ((lane >> 4) * 8);
    const bf16* aRow1 = aRow0 + (size_t)16 * lda;
    const bf16* bRow0 = Bt + (size_t)(nBase + (lane & 15)) * ldb + ((lane >> 4) * 8);
    const bf16* bRow1 = bRow0 + (size_t)16 * ldb;
    const bf16* bRow2 = bRow0 + (size_t)32 * ldb;
    const bf16* bRow3 = bRow0 + (size_t)48 * ldb;

    v8f acc[2][4];
#pragma unroll
    for (int mi = 0; mi < 2; ++mi)
#pragma unroll
        for (int ni = 0; ni < 4; ++ni)
#pragma unroll
            for (int r = 0; r < 8; ++r) acc[mi][ni][r] = 0.f;

    for (int k = 0; k < K; k += 32) {
        v16bf a0 = ldfragA(aRow0 + k);
        v16bf a1 = ldfragA(aRow1 + k);
        v16bf b0 = ldfragB(bRow0 + k);
        v16bf b1 = ldfragB(bRow1 + k);
        v16bf b2 = ldfragB(bRow2 + k);
        v16bf b3 = ldfragB(bRow3 + k);
        acc[0][0] = __builtin_amdgcn_wmma_f32_16x16x32_bf16(false, a0, false, b0, (short)0, acc[0][0], false, false);
        acc[0][1] = __builtin_amdgcn_wmma_f32_16x16x32_bf16(false, a0, false, b1, (short)0, acc[0][1], false, false);
        acc[0][2] = __builtin_amdgcn_wmma_f32_16x16x32_bf16(false, a0, false, b2, (short)0, acc[0][2], false, false);
        acc[0][3] = __builtin_amdgcn_wmma_f32_16x16x32_bf16(false, a0, false, b3, (short)0, acc[0][3], false, false);
        acc[1][0] = __builtin_amdgcn_wmma_f32_16x16x32_bf16(false, a1, false, b0, (short)0, acc[1][0], false, false);
        acc[1][1] = __builtin_amdgcn_wmma_f32_16x16x32_bf16(false, a1, false, b1, (short)0, acc[1][1], false, false);
        acc[1][2] = __builtin_amdgcn_wmma_f32_16x16x32_bf16(false, a1, false, b2, (short)0, acc[1][2], false, false);
        acc[1][3] = __builtin_amdgcn_wmma_f32_16x16x32_bf16(false, a1, false, b3, (short)0, acc[1][3], false, false);
    }

#pragma unroll
    for (int mi = 0; mi < 2; ++mi) {
#pragma unroll
        for (int r = 0; r < 8; ++r) {
            const int mm = mBase + mi * 16 + 8 * (lane >> 4) + r;
            const float* pp  = pose  ? pose  + (size_t)(mm % posT) * Nn : nullptr;
            const float* rp  = resid ? resid + (size_t)mm * Nn         : nullptr;
            float*       opF = outF  ? outF  + (size_t)mm * Nn         : nullptr;
            bf16*        opB = outB  ? outB  + (size_t)mm * Nn         : nullptr;
#pragma unroll
            for (int ni = 0; ni < 4; ++ni) {
                const int n = nBase + ni * 16 + (lane & 15);
                float v = acc[mi][ni][r];
                if (bias)    v += bias[n];
                if (pp)      v += pp[n];
                if (rp)      v += rp[n];
                if (do_gelu) v = gelu_exact(v);
                if (opF) opF[n] = v;
                if (opB) opB[n] = (bf16)v;
            }
        }
    }
}

// ---------------------------------------------------------------------------
// Attention: per (batch*R, head). S=77, dh=64. K/V staged in LDS, 2-pass softmax.
// Writes attention output directly as bf16 for the out-proj WMMA GEMM.
// ---------------------------------------------------------------------------
__global__ void attn_kernel(const float* __restrict__ qkv, bf16* __restrict__ ob) {
    const int br = blockIdx.x, h = blockIdx.y;
    __shared__ float Ks[TCLIPC * DHC];
    __shared__ float Vs[TCLIPC * DHC];
    const int tid = threadIdx.x;
    for (int e = tid; e < TCLIPC * DHC; e += blockDim.x) {
        int r = e >> 6, c = e & 63;
        size_t base = ((size_t)(br * TCLIPC + r)) * (3 * DC) + h * DHC + c;
        Ks[e] = qkv[base + DC];
        Vs[e] = qkv[base + 2 * DC];
    }
    __syncthreads();
    const int i = tid;
    if (i < TCLIPC) {
        float qr[DHC];
        const float* qp = qkv + ((size_t)(br * TCLIPC + i)) * (3 * DC) + h * DHC;
#pragma unroll
        for (int c = 0; c < DHC; ++c) qr[c] = qp[c];
        float mx = -INFINITY;
        for (int kk = 0; kk < TCLIPC; ++kk) {
            float s = 0.f;
            const float* kp = &Ks[kk * DHC];
#pragma unroll
            for (int c = 0; c < DHC; ++c) s += qr[c] * kp[c];
            mx = fmaxf(mx, s * 0.125f);
        }
        float ov[DHC];
#pragma unroll
        for (int c = 0; c < DHC; ++c) ov[c] = 0.f;
        float l = 0.f;
        for (int kk = 0; kk < TCLIPC; ++kk) {
            float s = 0.f;
            const float* kp = &Ks[kk * DHC];
#pragma unroll
            for (int c = 0; c < DHC; ++c) s += qr[c] * kp[c];
            float pexp = expf(s * 0.125f - mx);
            l += pexp;
            const float* vp = &Vs[kk * DHC];
#pragma unroll
            for (int c = 0; c < DHC; ++c) ov[c] += pexp * vp[c];
        }
        float inv = 1.f / l;
        bf16* op = ob + ((size_t)(br * TCLIPC + i)) * DC + h * DHC;
#pragma unroll
        for (int c = 0; c < DHC; ++c) op[c] = (bf16)(ov[c] * inv);
    }
}

// ---------------------------------------------------------------------------
// LayerNorm over D=512 (one row per block, 256 threads, 2 elems/thread)
// ---------------------------------------------------------------------------
__global__ void ln_kernel(const float* __restrict__ y, const float* __restrict__ g,
                          const float* __restrict__ beta, float* __restrict__ tF,
                          bf16* __restrict__ tB) {
    const int row = blockIdx.x, tid = threadIdx.x;
    const float* p = y + (size_t)row * DC;
    __shared__ float red[256];
    float a = p[tid], b = p[tid + 256];
    red[tid] = a + b; __syncthreads();
    for (int off = 128; off > 0; off >>= 1) { if (tid < off) red[tid] += red[tid + off]; __syncthreads(); }
    float mean = red[0] * (1.0f / DC);
    __syncthreads();
    float da = a - mean, db = b - mean;
    red[tid] = da * da + db * db; __syncthreads();
    for (int off = 128; off > 0; off >>= 1) { if (tid < off) red[tid] += red[tid + off]; __syncthreads(); }
    float inv = rsqrtf(red[0] * (1.0f / DC) + 1e-5f);
    float o1 = da * inv * g[tid]       + beta[tid];
    float o2 = db * inv * g[tid + 256] + beta[tid + 256];
    tF[(size_t)row * DC + tid]       = o1;
    tF[(size_t)row * DC + tid + 256] = o2;
    tB[(size_t)row * DC + tid]       = (bf16)o1;
    tB[(size_t)row * DC + tid + 256] = (bf16)o2;
}

// ---------------------------------------------------------------------------
// Output assembly: [B,R,50,512] = concat(text last token, motion[::4])
// ---------------------------------------------------------------------------
__global__ void output_kernel(const float* __restrict__ t, const float* __restrict__ x,
                              float* __restrict__ out) {
    const int g = blockIdx.x;             // br*50 + row
    const int br = g / 50, row = g % 50;
    const float* src = (row == 0)
        ? (t + ((size_t)(br * TCLIPC + (TCLIPC - 1))) * DC)
        : (x + ((size_t)(br * TMAXC + (row - 1) * 4)) * DC);
    float* dst = out + (size_t)g * DC;
    for (int c = threadIdx.x; c < DC; c += blockDim.x) dst[c] = src[c];
}

// ---------------------------------------------------------------------------
// Host: orchestrate the pipeline on `stream` (graph-capture safe)
// ---------------------------------------------------------------------------
extern "C" void kernel_launch(void* const* d_in, const int* in_sizes, int n_in,
                              void* d_out, int out_size, void* d_ws, size_t ws_size,
                              hipStream_t stream) {
    (void)in_sizes; (void)n_in; (void)out_size; (void)ws_size;
    const float* query_text    = (const float*)d_in[0];
    const float* db_text       = (const float*)d_in[1];
    const float* db_motions    = (const float*)d_in[2];
    const float* db_clipseq    = (const float*)d_in[3];
    const int*   lengths       = (const int*)  d_in[4];
    const int*   db_mlen       = (const int*)  d_in[5];
    const float* motion_proj_w = (const float*)d_in[6];
    const float* motion_proj_b = (const float*)d_in[7];
    const float* pos_emb       = (const float*)d_in[8];
    const float* ffn_w1        = (const float*)d_in[9];
    const float* ffn_b1        = (const float*)d_in[10];
    const float* ffn_w2        = (const float*)d_in[11];
    const float* ffn_b2        = (const float*)d_in[12];
    const float* te_wqkv       = (const float*)d_in[13];
    const float* te_bqkv       = (const float*)d_in[14];
    const float* te_wo         = (const float*)d_in[15];
    const float* te_bo         = (const float*)d_in[16];
    const float* te_ln1_g      = (const float*)d_in[17];
    const float* te_ln1_b      = (const float*)d_in[18];
    const float* te_ln2_g      = (const float*)d_in[19];
    const float* te_ln2_b      = (const float*)d_in[20];
    const float* te_w1         = (const float*)d_in[21];
    const float* te_b1         = (const float*)d_in[22];
    const float* te_w2         = (const float*)d_in[23];
    const float* te_b2         = (const float*)d_in[24];

    // Bump allocator over workspace (256B aligned)
    char* p = (char*)d_ws;
    auto alloc = [&](size_t bytes) -> void* {
        char* r = p; p += (bytes + 255) & ~(size_t)255; return (void*)r;
    };
    float* scores = (float*)alloc((size_t)BQ * NDB * 4);
    int*   idx    = (int*)  alloc((size_t)BRC * 4);
    bf16*  Wmt    = (bf16*) alloc((size_t)DC * DMP * 2);
    bf16*  W1t    = (bf16*) alloc((size_t)4 * DFFC * DC * 2);
    bf16*  W2t    = (bf16*) alloc((size_t)4 * DC * DFFC * 2);
    bf16*  Qkvt   = (bf16*) alloc((size_t)2 * (3 * DC) * DC * 2);
    bf16*  Wot    = (bf16*) alloc((size_t)2 * DC * DC * 2);
    bf16*  Tw1t   = (bf16*) alloc((size_t)2 * DFFC * DC * 2);
    bf16*  Tw2t   = (bf16*) alloc((size_t)2 * DC * DFFC * 2);
    bf16*  Xm     = (bf16*) alloc((size_t)M1C * DMP * 2);
    float* x      = (float*)alloc((size_t)M1C * DC * 4);
    bf16*  xb     = (bf16*) alloc((size_t)M1C * DC * 2);
    bf16*  hb     = (bf16*) alloc((size_t)M1C * DFFC * 2);   // shared motion/text hidden
    float* t      = (float*)alloc((size_t)M2C * DC * 4);
    bf16*  tb     = (bf16*) alloc((size_t)M2C * DC * 2);
    float* qkvb   = (float*)alloc((size_t)M2C * (3 * DC) * 4);
    bf16*  ob     = (bf16*) alloc((size_t)M2C * DC * 2);
    float* y      = (float*)alloc((size_t)M2C * DC * 4);

    // 1. retrieval scores + top-4
    score_kernel<<<dim3(BQ, NDB / 256), 256, 0, stream>>>(query_text, db_text, lengths, db_mlen, scores);
    topk_kernel<<<BQ, 256, 0, stream>>>(scores, idx);

    // 2. weight prep (f32 -> bf16 transposed)
    auto cvt = [&](const float* W, bf16* Wt, int K, int Nn, int Kp) {
        long total = (long)Nn * Kp;
        int blocks = (int)((total + 255) / 256); if (blocks > 2048) blocks = 2048;
        convw_kernel<<<blocks, 256, 0, stream>>>(W, Wt, K, Nn, Kp);
    };
    cvt(motion_proj_w, Wmt, DMC, DC, DMP);
    for (int i = 0; i < 4; ++i) cvt(ffn_w1 + (size_t)i * DC * DFFC, W1t + (size_t)i * DFFC * DC, DC, DFFC, DC);
    for (int i = 0; i < 4; ++i) cvt(ffn_w2 + (size_t)i * DFFC * DC, W2t + (size_t)i * DC * DFFC, DFFC, DC, DFFC);
    for (int i = 0; i < 2; ++i) cvt(te_wqkv + (size_t)i * DC * (3*DC), Qkvt + (size_t)i * (3*DC) * DC, DC, 3*DC, DC);
    for (int i = 0; i < 2; ++i) cvt(te_wo   + (size_t)i * DC * DC,     Wot  + (size_t)i * DC * DC,     DC, DC, DC);
    for (int i = 0; i < 2; ++i) cvt(te_w1   + (size_t)i * DC * DFFC,   Tw1t + (size_t)i * DFFC * DC,   DC, DFFC, DC);
    for (int i = 0; i < 2; ++i) cvt(te_w2   + (size_t)i * DFFC * DC,   Tw2t + (size_t)i * DC * DFFC,   DFFC, DC, DFFC);

    auto gemm = [&](const bf16* A, int lda, const bf16* Bt, int ldb, const float* bias,
                    const float* resid, const float* pose, int posT,
                    float* oF, bf16* oB, int M, int Nn, int K, int gel) {
        gemm_bf16_kernel<<<dim3(M / 128, Nn / 128), 256, 0, stream>>>(
            A, lda, Bt, ldb, bias, resid, pose, posT, oF, oB, M, Nn, K, gel);
    };

    // 3. motion branch: gather -> proj(+bias+pos) -> 4x residual FFN
    gather_motion_kernel<<<M1C, 128, 0, stream>>>(db_motions, idx, Xm);
    gemm(Xm, DMP, Wmt, DMP, motion_proj_b, nullptr, pos_emb, TMAXC, x, xb, M1C, DC, DMP, 0);
    for (int i = 0; i < 4; ++i) {
        gemm(xb, DC,  W1t + (size_t)i * DFFC * DC, DC,  ffn_b1 + (size_t)i * DFFC, nullptr, nullptr, 0,
             nullptr, hb, M1C, DFFC, DC, 1);
        gemm(hb, DFFC, W2t + (size_t)i * DC * DFFC, DFFC, ffn_b2 + (size_t)i * DC, x, nullptr, 0,
             x, xb, M1C, DC, DFFC, 0);
    }

    // 4. text branch: gather -> 2x post-norm transformer layers
    gather_text_kernel<<<M2C, 128, 0, stream>>>(db_clipseq, idx, t, tb);
    for (int i = 0; i < 2; ++i) {
        gemm(tb, DC, Qkvt + (size_t)i * (3*DC) * DC, DC, te_bqkv + (size_t)i * (3*DC), nullptr, nullptr, 0,
             qkvb, nullptr, M2C, 3 * DC, DC, 0);
        attn_kernel<<<dim3(BRC, HC), 128, 0, stream>>>(qkvb, ob);
        gemm(ob, DC, Wot + (size_t)i * DC * DC, DC, te_bo + (size_t)i * DC, t, nullptr, 0,
             y, nullptr, M2C, DC, DC, 0);
        ln_kernel<<<M2C, 256, 0, stream>>>(y, te_ln1_g + (size_t)i * DC, te_ln1_b + (size_t)i * DC, t, tb);
        gemm(tb, DC,  Tw1t + (size_t)i * DFFC * DC, DC,  te_b1 + (size_t)i * DFFC, nullptr, nullptr, 0,
             nullptr, hb, M2C, DFFC, DC, 1);
        gemm(hb, DFFC, Tw2t + (size_t)i * DC * DFFC, DFFC, te_b2 + (size_t)i * DC, t, nullptr, 0,
             y, nullptr, M2C, DC, DFFC, 0);
        ln_kernel<<<M2C, 256, 0, stream>>>(y, te_ln2_g + (size_t)i * DC, te_ln2_b + (size_t)i * DC, t, tb);
    }

    // 5. assemble output [B,R,50,512]
    output_kernel<<<BRC * 50, 256, 0, stream>>>(t, x, (float*)d_out);
}
